// Sampler_63960652972361
// MI455X (gfx1250) — matrix operand order repaired and checked by reading, not compiled, not run on hardware
//
#include <hip/hip_runtime.h>
#include <hip/hip_bf16.h>
#include <math.h>

typedef __attribute__((ext_vector_type(16))) _Float16 v16h;
typedef __attribute__((ext_vector_type(8)))  _Float16 v8h;
typedef __attribute__((ext_vector_type(4)))  _Float16 v4h;
typedef __attribute__((ext_vector_type(8)))  float    v8f;
typedef __attribute__((ext_vector_type(4)))  float    v4f;

#define B_ 64
#define D_ 512
#define S_ 2048
#define K_ 8
#define E_ 1024   // 2*D

// ---------------- mean over S of x[b,d,:] -> info_g[b*D+d] ----------------
__global__ void __launch_bounds__(256)
mean_kernel(const float* __restrict__ x, float* __restrict__ info_g) {
    __shared__ float red[256];
    int row = blockIdx.x;                       // b*D + d
    const float* p = x + (size_t)row * S_;
    float s = 0.f;
    for (int i = threadIdx.x; i < S_; i += 256) s += p[i];
    red[threadIdx.x] = s; __syncthreads();
    for (int off = 128; off > 0; off >>= 1) {
        if (threadIdx.x < off) red[threadIdx.x] += red[threadIdx.x + off];
        __syncthreads();
    }
    if (threadIdx.x == 0) info_g[row] = red[0] * (1.0f / (float)S_);
}

// ---------------- W1 (f32, [D, 2D] row-major) -> f16 ----------------
__global__ void __launch_bounds__(256)
w1_kernel(const float* __restrict__ W1, _Float16* __restrict__ W1h) {
    int i = blockIdx.x * 256 + threadIdx.x;
    W1h[i] = (_Float16)W1[i];
}

// ---------------- sinusoidal PE table [S, 2D] f32 ----------------
__global__ void __launch_bounds__(256)
pe_kernel(float* __restrict__ PE) {
    int i = blockIdx.x * 256 + threadIdx.x;     // s*E + e
    int s = i >> 10, e = i & (E_ - 1);
    int j = e >> 1;
    float ang = (float)s * __expf(-9.210340371976184f * (float)(2 * j) * (1.0f / 1024.0f));
    PE[i] = (e & 1) ? cosf(ang) : sinf(ang);
}

// ---------------- transpose x[B,D,S] -> feature[B,S,D] (output #3) ----------------
__global__ void __launch_bounds__(256)
transpose_kernel(const float* __restrict__ x, float* __restrict__ feat) {
    __shared__ float tile[32][33];
    int b = blockIdx.z;
    int s0 = blockIdx.x * 32, d0 = blockIdx.y * 32;
    const float* xb = x + (size_t)b * D_ * S_;
    float* fb = feat + (size_t)b * S_ * D_;
    #pragma unroll
    for (int k = 0; k < 32; k += 8)
        tile[threadIdx.y + k][threadIdx.x] =
            xb[(size_t)(d0 + threadIdx.y + k) * S_ + s0 + threadIdx.x];
    __syncthreads();
    #pragma unroll
    for (int k = 0; k < 32; k += 8)
        fb[(size_t)(s0 + threadIdx.y + k) * D_ + d0 + threadIdx.x] =
            tile[threadIdx.x][threadIdx.y + k];
}

// ---------------- fused GEMM (WMMA f16->f32) + bias + ReLU + W2 dot -> score ----------------
// h[b,s,d] = relu( sum_e (xc+PE)[b,s,e] * W1[d,e] + b1[d] );  score[b,s] = sum_d h*W2[d]
#define A_STRIDE 1032   // 1024 + 8 halves: breaks 64-bank LDS aliasing on row-strided b128 loads

__device__ __forceinline__ v16h frag16(const _Float16* lo, const _Float16* hi) {
    v8h l = *(const v8h*)lo;
    v8h h = *(const v8h*)hi;
    v16h r;
    #pragma unroll
    for (int i = 0; i < 8; ++i) { r[i] = l[i]; r[i + 8] = h[i]; }
    return r;
}

__global__ void __launch_bounds__(256)
gemm_score_kernel(const float* __restrict__ feat, const float* __restrict__ info_g,
                  const _Float16* __restrict__ W1h, const float* __restrict__ PE,
                  const float* __restrict__ b1, const float* __restrict__ W2,
                  float* __restrict__ score) {
    __shared__ _Float16 aT[16 * A_STRIDE];
    __shared__ float sScore[16];
    int b   = blockIdx.y;
    int m0  = blockIdx.x * 16;
    int tid = threadIdx.x;

    // stage A: 16 rows of (xc + PE) as f16 in LDS, vectorized 4-wide
    const float* fb = feat + ((size_t)b * S_ + m0) * D_;
    const float* ig = info_g + (size_t)b * D_;
    const float* pe = PE + (size_t)m0 * E_;
    for (int i = tid; i < 16 * (E_ / 4); i += 256) {
        int r = i >> 8, e = (i & 255) * 4;           // e multiple of 4; D_ boundary aligned
        v4f xv = (e < D_) ? *(const v4f*)(fb + (size_t)r * D_ + e)
                          : *(const v4f*)(ig + (e - D_));
        v4f pv = *(const v4f*)(pe + (size_t)r * E_ + e);
        v4h out;
        #pragma unroll
        for (int q = 0; q < 4; ++q) out[q] = (_Float16)(xv[q] + pv[q]);
        *(v4h*)(aT + r * A_STRIDE + e) = out;
    }
    if (tid < 16) sScore[tid] = 0.0f;
    __syncthreads();

    int wave = tid >> 5;
    int lane = tid & 31;
    int half = lane >> 4;   // which 16-lane half
    int r    = lane & 15;   // A: row M; B/C/D: column N

    // each wave owns 4 consecutive N-tiles: columns wave*64 .. wave*64+63
    const _Float16* aRow = aT + r * A_STRIDE + half * 8;     // ISA A-layout base
    const _Float16* wrow[4];
    float b1v[4], w2v[4];
    #pragma unroll
    for (int t = 0; t < 4; ++t) {
        int nCol = wave * 64 + t * 16 + r;
        wrow[t] = W1h + (size_t)nCol * E_ + half * 8;        // B: ISA layout matching A (K = kk + half*8 + e, +16 for upper 8), N = r
        b1v[t] = b1[nCol];
        w2v[t] = W2[nCol];
    }

    v8f c[4] = {};
    // ping-pong double-buffered software pipeline (no register-rotation movs)
    v16h aA = frag16(aRow, aRow + 16);
    v16h bA[4];
    #pragma unroll
    for (int t = 0; t < 4; ++t) bA[t] = frag16(wrow[t], wrow[t] + 16);

    for (int kk = 0; kk < E_; kk += 64) {
        int k1 = kk + 32;
        v16h aB = frag16(aRow + k1, aRow + k1 + 16);
        v16h bB[4];
        #pragma unroll
        for (int t = 0; t < 4; ++t) bB[t] = frag16(wrow[t] + k1, wrow[t] + k1 + 16);
        #pragma unroll
        for (int t = 0; t < 4; ++t)
            c[t] = __builtin_amdgcn_wmma_f32_16x16x32_f16(
                       false, aA, false, bA[t], (short)0, c[t], false, false);

        int k2 = (kk + 64) & (E_ - 1);               // wraps: dead-but-safe final prefetch
        aA = frag16(aRow + k2, aRow + k2 + 16);
        #pragma unroll
        for (int t = 0; t < 4; ++t) bA[t] = frag16(wrow[t] + k2, wrow[t] + k2 + 16);
        #pragma unroll
        for (int t = 0; t < 4; ++t)
            c[t] = __builtin_amdgcn_wmma_f32_16x16x32_f16(
                       false, aB, false, bB[t], (short)0, c[t], false, false);
    }

    // epilogue: relu(c + b1[n]) * W2[n]; C layout: lane=N(=r), VGPR v -> M = v + 8*half
    #pragma unroll
    for (int t = 0; t < 4; ++t) {
        #pragma unroll
        for (int v = 0; v < 8; ++v) {
            float u = fmaxf(c[t][v] + b1v[t], 0.0f) * w2v[t];
            u += __shfl_xor(u, 1, 32);
            u += __shfl_xor(u, 2, 32);
            u += __shfl_xor(u, 4, 32);
            u += __shfl_xor(u, 8, 32);   // xor<=8 stays within the half
            if (r == 0) atomicAdd(&sScore[v + half * 8], u);
        }
    }
    __syncthreads();
    if (tid < 16) score[(size_t)b * S_ + m0 + tid] = sScore[tid];
    // note: +b2 and min-max normalization are monotone -> irrelevant for top-k
}

// ---------------- per-batch top-8, ascending sort, one-hot, gather ----------------
__global__ void __launch_bounds__(256)
topk_kernel(const float* __restrict__ score, const float* __restrict__ feat,
            float* __restrict__ out_idx, float* __restrict__ out_sel) {
    __shared__ float sv[S_];
    __shared__ float bv[256];
    __shared__ int   bi[256];
    __shared__ int   kIdx[K_];
    int b = blockIdx.x, tid = threadIdx.x;
    const float* sc = score + (size_t)b * S_;
    for (int i = tid; i < S_; i += 256) sv[i] = sc[i];
    __syncthreads();

    for (int k = 0; k < K_; ++k) {
        float best = -__builtin_inff(); int bidx = S_;
        for (int i = tid; i < S_; i += 256) {
            float v = sv[i];
            if (v > best || (v == best && i < bidx)) { best = v; bidx = i; }
        }
        bv[tid] = best; bi[tid] = bidx;
        __syncthreads();
        for (int off = 128; off > 0; off >>= 1) {
            if (tid < off) {
                float v2 = bv[tid + off]; int i2 = bi[tid + off];
                if (v2 > bv[tid] || (v2 == bv[tid] && i2 < bi[tid])) { bv[tid] = v2; bi[tid] = i2; }
            }
            __syncthreads();
        }
        if (tid == 0) { kIdx[k] = bi[0]; sv[bi[0]] = -__builtin_inff(); }
        __syncthreads();
    }
    if (tid == 0) {   // insertion sort ascending (jax sorts selected indices)
        for (int i = 1; i < K_; ++i) {
            int key = kIdx[i], j = i - 1;
            while (j >= 0 && kIdx[j] > key) { kIdx[j + 1] = kIdx[j]; --j; }
            kIdx[j + 1] = key;
        }
    }
    __syncthreads();

    float* oi = out_idx + (size_t)b * K_ * S_;
    for (int i = tid; i < K_ * S_; i += 256) {
        int k = i >> 11, s = i & (S_ - 1);
        oi[i] = (s == kIdx[k]) ? 1.0f : 0.0f;
    }
    float* os = out_sel + (size_t)b * K_ * D_;
    const float* fbp = feat + (size_t)b * S_ * D_;
    for (int i = tid; i < K_ * D_; i += 256) {
        int k = i >> 9, d = i & (D_ - 1);
        os[i] = fbp[(size_t)kIdx[k] * D_ + d];
    }
}

extern "C" void kernel_launch(void* const* d_in, const int* in_sizes, int n_in,
                              void* d_out, int out_size, void* d_ws, size_t ws_size,
                              hipStream_t stream) {
    const float* x  = (const float*)d_in[0];   // [B, D, S]
    const float* W1 = (const float*)d_in[1];   // [D, 2D]
    const float* b1 = (const float*)d_in[2];   // [D]
    const float* W2 = (const float*)d_in[3];   // [1, D]
    // d_in[4] = b2: constant shift, does not affect top-k; not needed.

    // workspace layout (~10 MB)
    char* ws = (char*)d_ws;
    float*    info_g = (float*)ws;                                   // B*D f32   (128 KB)
    _Float16* W1h    = (_Float16*)(ws + 131072);                     // D*E f16   (1 MB)
    float*    PE     = (float*)(ws + 131072 + 1048576);              // S*E f32   (8 MB)
    float*    score  = (float*)(ws + 131072 + 1048576 + 8388608);    // B*S f32   (512 KB)

    // outputs, concatenated flat in reference return order
    float* out_idx = (float*)d_out;                          // indices [B,K,S]
    float* out_sel = out_idx + (size_t)B_ * K_ * S_;         // selected [B,K,D]
    float* feat    = out_sel + (size_t)B_ * K_ * D_;         // feature [B,S,D]

    mean_kernel<<<B_ * D_, 256, 0, stream>>>(x, info_g);
    w1_kernel<<<(D_ * E_) / 256, 256, 0, stream>>>(W1, W1h);
    pe_kernel<<<(S_ * E_) / 256, 256, 0, stream>>>(PE);
    transpose_kernel<<<dim3(S_ / 32, D_ / 32, B_), dim3(32, 8), 0, stream>>>(x, feat);
    gemm_score_kernel<<<dim3(S_ / 16, B_), 256, 0, stream>>>(feat, info_g, W1h, PE, b1, W2, score);
    topk_kernel<<<B_, 256, 0, stream>>>(score, feat, out_idx, out_sel);
}
